// DictField_59923383714484
// MI455X (gfx1250) — matrix operand order repaired and checked by reading, not compiled, not run on hardware
//
#include <hip/hip_runtime.h>

typedef __attribute__((ext_vector_type(16))) _Float16 v16h;
typedef __attribute__((ext_vector_type(8)))  float    v8f;

#define NPTS       (640 * 368)
#define PTS_BLK    128
#define NBLK       (NPTS / PTS_BLK)   // 1840 exactly
#define K0PAD      160                // 144 padded to 5*32
#define HID        256
#define LDX        K0PAD
#define LDH        HID

// d_ws layout (f16): W0t [256][160], W1t [256][256], W2t [16][256]
#define W0T_OFF    0
#define W1T_OFF    (256 * 160 * 2)
#define W2T_OFF    (W1T_OFF + 256 * 256 * 2)

union FragU {
    uint4 q[2];
    v16h  h;
};

// A-fragment (16x32 f16) from row-major [row][k] buffer, stride in halves.
// Lane<16: row M=lane, halves = K[k0..k0+7] then K[k0+16..k0+23].
// Lane>=16: row M=lane-16, halves = K[k0+8..k0+15] then K[k0+24..k0+31].
__device__ inline v16h load_a_frag(const _Float16* base, int row0, int k0, int stride) {
    int lane = threadIdx.x & 31;
    int r    = lane & 15;
    int kh   = (lane >> 4) << 3;     // 0 or 8
    const _Float16* src = base + (row0 + r) * stride + k0 + kh;
    FragU f;
    f.q[0] = *(const uint4*)(src);
    f.q[1] = *(const uint4*)(src + 16);
    return f.h;
}

// B-fragment (32x16 f16) from row-major transposed weights Wt[n][k].
// Lane<16: col N=lane, halves = K[k0..k0+7] then K[k0+16..k0+23]; lane>=16: col N=lane-16, K[k0+8..k0+15] then K[k0+24..k0+31].
__device__ inline v16h load_b_frag(const _Float16* base, int n0, int k0, int stride) {
    int lane = threadIdx.x & 31;
    int n    = lane & 15;
    int ko   = (lane >> 4) << 3;     // 0 or 8
    const _Float16* src = base + (n0 + n) * stride + k0 + ko;
    FragU f;
    f.q[0] = *(const uint4*)(src);
    f.q[1] = *(const uint4*)(src + 16);
    return f.h;
}

__device__ inline v8f wmma_f16(v16h a, v16h b, v8f c) {
    return __builtin_amdgcn_wmma_f32_16x16x32_f16(false, a, false, b, (short)0, c, false, false);
}

// ---------------- weight prep: fp32 -> f16, transpose, pad ----------------
__global__ void __launch_bounds__(256)
prep_weights(const float* __restrict__ w0, const float* __restrict__ w1,
             const float* __restrict__ w2, _Float16* __restrict__ W0t,
             _Float16* __restrict__ W1t, _Float16* __restrict__ W2t) {
    int i = blockIdx.x * blockDim.x + threadIdx.x;
    if (i < 256 * K0PAD) {           // W0t[n][k], k<144 valid
        int n = i / K0PAD, k = i - n * K0PAD;
        W0t[i] = (_Float16)(k < 144 ? w0[k * HID + n] : 0.0f);
    }
    if (i < 256 * 256) {             // W1t[n][k]
        int n = i >> 8, k = i & 255;
        W1t[i] = (_Float16)w1[k * HID + n];
    }
    if (i < 16 * 256) {              // W2t[n][k], n<2 valid
        int n = i >> 8, k = i & 255;
        W2t[i] = (_Float16)(n < 2 ? w2[k * 2 + n] : 0.0f);
    }
}

// ---------------- fused field + MLP kernel ----------------
__global__ void __launch_bounds__(256)
dictfield_kernel(const float* __restrict__ coords, const float* __restrict__ coeffs,
                 const float* __restrict__ B0, const float* __restrict__ B1,
                 const float* __restrict__ B2, const float* __restrict__ B3,
                 const float* __restrict__ B4, const float* __restrict__ B5,
                 const float* __restrict__ b0g, const float* __restrict__ b1g,
                 const _Float16* __restrict__ W0t, const _Float16* __restrict__ W1t,
                 const _Float16* __restrict__ W2t, float* __restrict__ out) {
    __shared__ _Float16 Xs[PTS_BLK * LDX];     // 40 KB feature tile (f16)
    __shared__ _Float16 Hs[PTS_BLK * LDH];     // 64 KB hidden tile  (f16)
    __shared__ int4   sidx[PTS_BLK][7];        // bilinear tap indices
    __shared__ float4 swt [PTS_BLK][7];        // bilinear tap weights

    const int tid  = threadIdx.x;
    const int lane = tid & 31;
    const int wid  = tid >> 5;                 // 8 waves, each owns 16 rows
    const int pbase = blockIdx.x * PTS_BLK;

    // ---- stage 0: per-point sample parameters (threads 0..127) ----
    if (tid < PTS_BLK) {
        const int p  = tid;
        const int gp = pbase + p;
        const float cx = coords[2 * gp + 0];
        const float cy = coords[2 * gp + 1];

        // coeff grid (align_corners = False, border clip), 64x64
        {
            float gx = cx * (2.0f / 640.0f) - 1.0f;
            float gy = cy * (2.0f / 368.0f) - 1.0f;
            float ix = fminf(fmaxf(((gx + 1.0f) * 64.0f - 1.0f) * 0.5f, 0.0f), 63.0f);
            float iy = fminf(fmaxf(((gy + 1.0f) * 64.0f - 1.0f) * 0.5f, 0.0f), 63.0f);
            float fx = floorf(ix), fy = floorf(iy);
            float wx = ix - fx,     wy = iy - fy;
            int x0 = (int)fx, y0 = (int)fy;
            int x1 = min(x0 + 1, 63), y1 = min(y0 + 1, 63);
            sidx[p][0] = make_int4(y0 * 64 + x0, y0 * 64 + x1, y1 * 64 + x0, y1 * 64 + x1);
            swt [p][0] = make_float4((1.0f - wy) * (1.0f - wx), (1.0f - wy) * wx,
                                     wy * (1.0f - wx), wy * wx);
        }
        // basis grids (align_corners = True)
        const int RESO[6] = {32, 51, 70, 89, 108, 128};
        #pragma unroll
        for (int b = 0; b < 6; ++b) {
            const int   R = RESO[b];
            const float s = (float)R;
            float lx = fminf(fmaxf(fmodf(cx, s) / (s * 0.5f) - 1.0f, -1.0f), 1.0f);
            float ly = fminf(fmaxf(fmodf(cy, s) / (s * 0.5f) - 1.0f, -1.0f), 1.0f);
            float ix = (lx + 1.0f) * 0.5f * (float)(R - 1);
            float iy = (ly + 1.0f) * 0.5f * (float)(R - 1);
            float fx = floorf(ix), fy = floorf(iy);
            float wx = ix - fx,     wy = iy - fy;
            int x0 = min(max((int)fx, 0), R - 1), y0 = min(max((int)fy, 0), R - 1);
            int x1 = min(x0 + 1, R - 1),          y1 = min(y0 + 1, R - 1);
            sidx[p][1 + b] = make_int4(y0 * R + x0, y0 * R + x1, y1 * R + x0, y1 * R + x1);
            swt [p][1 + b] = make_float4((1.0f - wy) * (1.0f - wx), (1.0f - wy) * wx,
                                         wy * (1.0f - wx), wy * wx);
        }
    }
    __syncthreads();

    // ---- stage 1: cooperative feature fill: X[p][c] = coeff(c) * basis(c) ----
    for (int it = tid; it < PTS_BLK * K0PAD; it += 256) {
        const int p = it / K0PAD;
        const int c = it - p * K0PAD;
        _Float16 val = (_Float16)0.0f;
        if (c < 144) {
            const float* bp; int R, start, slot;
            if (c < 96) {
                if (c < 32)      { bp = B0; R = 32;  start = 0;   slot = 1; }
                else if (c < 64) { bp = B1; R = 51;  start = 32;  slot = 2; }
                else             { bp = B2; R = 70;  start = 64;  slot = 3; }
            } else {
                if (c < 112)     { bp = B3; R = 89;  start = 96;  slot = 4; }
                else if (c < 128){ bp = B4; R = 108; start = 112; slot = 5; }
                else             { bp = B5; R = 128; start = 128; slot = 6; }
            }
            const int4   ci = sidx[p][0];
            const float4 cw = swt [p][0];
            const float* cp = coeffs + c * 4096;   // 64*64 per channel
            float cv = cp[ci.x] * cw.x + cp[ci.y] * cw.y + cp[ci.z] * cw.z + cp[ci.w] * cw.w;
            const int4   bi = sidx[p][slot];
            const float4 bw = swt [p][slot];
            const float* bb = bp + (c - start) * R * R;
            float bv = bb[bi.x] * bw.x + bb[bi.y] * bw.y + bb[bi.z] * bw.z + bb[bi.w] * bw.w;
            val = (_Float16)(cv * bv);
        }
        Xs[p * LDX + c] = val;
    }
    __syncthreads();

    const int row0 = wid * 16;        // this wave's M-tile inside the block

    // ---- layer 0: H = relu(X @ W0 + b0), K = 160 (5 wmma steps), N = 256 ----
    {
        v16h a[5];
        #pragma unroll
        for (int kk = 0; kk < 5; ++kk) a[kk] = load_a_frag(Xs, row0, kk * 32, LDX);
        for (int nt = 0; nt < 16; ++nt) {
            v8f acc = {};
            #pragma unroll
            for (int kk = 0; kk < 5; ++kk)
                acc = wmma_f16(a[kk], load_b_frag(W0t, nt * 16, kk * 32, K0PAD), acc);
            const float bias = b0g[nt * 16 + (lane & 15)];
            const int   col  = nt * 16 + (lane & 15);
            const int   rb   = row0 + ((lane >> 4) << 3);
            #pragma unroll
            for (int i = 0; i < 8; ++i)
                Hs[(rb + i) * LDH + col] = (_Float16)fmaxf(acc[i] + bias, 0.0f);
        }
    }
    __syncthreads();

    // ---- layer 1: H = relu(H @ W1 + b1), K = 256 (8 wmma steps), N = 256 ----
    {
        v16h a[8];
        #pragma unroll
        for (int kk = 0; kk < 8; ++kk) a[kk] = load_a_frag(Hs, row0, kk * 32, LDH);
        for (int nt = 0; nt < 16; ++nt) {
            v8f acc = {};
            #pragma unroll
            for (int kk = 0; kk < 8; ++kk)
                acc = wmma_f16(a[kk], load_b_frag(W1t, nt * 16, kk * 32, HID), acc);
            const float bias = b1g[nt * 16 + (lane & 15)];
            const int   col  = nt * 16 + (lane & 15);
            const int   rb   = row0 + ((lane >> 4) << 3);
            #pragma unroll
            for (int i = 0; i < 8; ++i)
                Hs[(rb + i) * LDH + col] = (_Float16)fmaxf(acc[i] + bias, 0.0f);
        }
    }
    __syncthreads();

    // ---- layer 2: out = H @ W2, K = 256, N = 16 (2 valid columns) ----
    {
        v16h a[8];
        #pragma unroll
        for (int kk = 0; kk < 8; ++kk) a[kk] = load_a_frag(Hs, row0, kk * 32, LDH);
        v8f acc = {};
        #pragma unroll
        for (int kk = 0; kk < 8; ++kk)
            acc = wmma_f16(a[kk], load_b_frag(W2t, 0, kk * 32, HID), acc);
        const int n = lane & 15;
        if (n < 2) {
            const int rb = pbase + row0 + ((lane >> 4) << 3);
            #pragma unroll
            for (int i = 0; i < 8; ++i)
                out[(rb + i) * 2 + n] = acc[i];
        }
    }
}

extern "C" void kernel_launch(void* const* d_in, const int* in_sizes, int n_in,
                              void* d_out, int out_size, void* d_ws, size_t ws_size,
                              hipStream_t stream) {
    const float* coords = (const float*)d_in[0];
    const float* coeffs = (const float*)d_in[1];
    const float* B0 = (const float*)d_in[2];
    const float* B1 = (const float*)d_in[3];
    const float* B2 = (const float*)d_in[4];
    const float* B3 = (const float*)d_in[5];
    const float* B4 = (const float*)d_in[6];
    const float* B5 = (const float*)d_in[7];
    const float* w0 = (const float*)d_in[8];
    const float* b0 = (const float*)d_in[9];
    const float* w1 = (const float*)d_in[10];
    const float* b1 = (const float*)d_in[11];
    const float* w2 = (const float*)d_in[12];
    float* out = (float*)d_out;

    _Float16* W0t = (_Float16*)((char*)d_ws + W0T_OFF);
    _Float16* W1t = (_Float16*)((char*)d_ws + W1T_OFF);
    _Float16* W2t = (_Float16*)((char*)d_ws + W2T_OFF);

    prep_weights<<<256, 256, 0, stream>>>(w0, w1, w2, W0t, W1t, W2t);
    dictfield_kernel<<<NBLK, 256, 0, stream>>>(coords, coeffs, B0, B1, B2, B3, B4, B5,
                                               b0, b1, W0t, W1t, W2t, out);
}
